// RingAttentionCorrect_86517821216464
// MI455X (gfx1250) — matrix operand order repaired and checked by reading, not compiled, not run on hardware
//
#include <hip/hip_runtime.h>

typedef __attribute__((ext_vector_type(16))) __bf16 v16bf;
typedef __attribute__((ext_vector_type(8)))  __bf16 v8bf;
typedef __attribute__((ext_vector_type(8)))  float  v8f;

namespace {
constexpr int kN     = 4096;
constexpr int kH     = 16;
constexpr int kD     = 64;
constexpr int kRing  = 4;
constexpr int kChunk = kN / kRing;     // 1024 keys per ring chunk
constexpr int kKVT   = 64;             // keys per LDS tile
constexpr int kWaves = 8;              // waves per workgroup
constexpr int kQPB   = kWaves * 16;    // 128 query rows per workgroup
constexpr int kTiles = kChunk / kKVT;  // 16 tiles per chunk
// LDS row stride: 72 elements = 144 B = 36 dwords. 36*l16 mod 64 hits 16 distinct
// banks -> conflict-free b128 operand loads; 144 B keeps rows 16-byte aligned.
constexpr int kLds   = 72;
}

// Load 16 contiguous bf16 (two 16-byte chunks) as a v16bf WMMA operand.
__device__ __forceinline__ v16bf ld16(const __bf16* p) {
  union { v16bf v; v8bf h[2]; } u;
  u.h[0] = *(const v8bf*)p;
  u.h[1] = *(const v8bf*)(p + 8);
  return u.v;
}

__global__ __launch_bounds__(kWaves * 32)
void ring_attn_bf16_wmma(const float* __restrict__ Qg,
                         const float* __restrict__ Kg,
                         const float* __restrict__ Vg,
                         float* __restrict__ Og)
{
  // K tile row-major [key][dim], V tile transposed [dim][key], per-wave P staging.
  __shared__ __align__(16) __bf16 sK [kKVT][kLds];        // ~9 KB
  __shared__ __align__(16) __bf16 sVt[kD][kLds];          // ~9 KB
  __shared__ __align__(16) __bf16 sP [kWaves][16][kLds];  // ~18 KB

  const int h    = blockIdx.y;
  const int qb   = blockIdx.x * kQPB;
  const int tid  = threadIdx.x;
  const int wave = tid >> 5;
  const int lane = tid & 31;
  const int half = lane >> 4;   // which 16-lane half
  const int l16  = lane & 15;

  // ---- Q A-fragments once; fold 1/sqrt(D) * log2(e) so softmax uses exp2 ----
  const float qscale = 0.125f * 1.44269504088896340736f;
  const int qrow = qb + wave * 16 + l16;                // A layout: lane row M = lane&15
  v16bf qf[2];
#pragma unroll
  for (int kk = 0; kk < 2; ++kk) {
    const float* qp = Qg + ((size_t)qrow * kH + h) * kD + kk * 32;
#pragma unroll
    for (int e = 0; e < 8; ++e) {
      // A-frag k order used here (B matches): elems 0..15 -> K = 16*half + elem (contiguous)
      qf[kk][e]     = (__bf16)(qp[16 * half + e] * qscale);
      qf[kk][e + 8] = (__bf16)(qp[16 * half + 8 + e] * qscale);
    }
  }

  v8f acc[4];
#pragma unroll
  for (int ng = 0; ng < 4; ++ng)
#pragma unroll
    for (int i = 0; i < 8; ++i) acc[ng][i] = 0.0f;

  // K stager: thread -> one key row, 16-dim slice (two b128 LDS stores).
  const int kRowId = tid >> 2;         // 0..63
  const int kCol0  = (tid & 3) * 16;   // 0,16,32,48
  // V stager: thread -> 4x4 sub-tile, written as 4 packed b64 column stores.
  const int vRow0  = (tid >> 4) * 4;   // kv base (0..60)
  const int vCol0  = (tid & 15) * 4;   // dim base (0..60)

  for (int r = 0; r < kRing; ++r) {
    // per-chunk online-softmax state (independent softmax per chunk, per reference)
    float m[8], lsum[8];
    v8f o[4];
#pragma unroll
    for (int g = 0; g < 8; ++g) { m[g] = -1e30f; lsum[g] = 0.0f; }
#pragma unroll
    for (int ng = 0; ng < 4; ++ng)
#pragma unroll
      for (int i = 0; i < 8; ++i) o[ng][i] = 0.0f;

    for (int kt = 0; kt < kTiles; ++kt) {
      const int kv0 = r * kChunk + kt * kKVT;
      __syncthreads();
      {
        // ---- stage K: convert f32 -> bf16, two 16-byte vector stores ----
        const float* kp = Kg + ((size_t)(kv0 + kRowId) * kH + h) * kD + kCol0;
        __builtin_prefetch(kp + (size_t)kKVT * kH * kD, 0, 3);  // next tile
        v8bf kb[2];
#pragma unroll
        for (int e = 0; e < 8; ++e) {
          kb[0][e] = (__bf16)kp[e];
          kb[1][e] = (__bf16)kp[8 + e];
        }
        *(v8bf*)&sK[kRowId][kCol0]     = kb[0];
        *(v8bf*)&sK[kRowId][kCol0 + 8] = kb[1];

        // ---- stage V transposed: 4x4 sub-tile, 4 packed b64 column stores ----
        float vv[4][4];
#pragma unroll
        for (int rr = 0; rr < 4; ++rr) {
          const float* vp = Vg + ((size_t)(kv0 + vRow0 + rr) * kH + h) * kD + vCol0;
          if (rr == 0) __builtin_prefetch(vp + (size_t)kKVT * kH * kD, 0, 3);
#pragma unroll
          for (int cc = 0; cc < 4; ++cc) vv[rr][cc] = vp[cc];
        }
#pragma unroll
        for (int cc = 0; cc < 4; ++cc) {
          union { __bf16 e[4]; uint2 u; } pk;
#pragma unroll
          for (int rr = 0; rr < 4; ++rr) pk.e[rr] = (__bf16)vv[rr][cc];
          *(uint2*)&sVt[vCol0 + cc][vRow0] = pk.u;   // 8-byte aligned (vRow0 % 4 == 0)
        }
      }
      __syncthreads();

      // ---- S = Q * K^T (log2 domain), 16x64 scores in four 16x16 C-frags ----
      v8f S[4];
#pragma unroll
      for (int n = 0; n < 4; ++n)
#pragma unroll
        for (int i = 0; i < 8; ++i) S[n][i] = 0.0f;
#pragma unroll
      for (int n = 0; n < 4; ++n)
#pragma unroll
        for (int kk = 0; kk < 2; ++kk) {
          // B-frag: lane column N = l16 (key n*16+l16); K-run = kk*32 + 16*half .. +15
          v16bf b = ld16(&sK[n * 16 + l16][kk * 32 + 16 * half]);
          S[n] = __builtin_amdgcn_wmma_f32_16x16x32_bf16(false, qf[kk], false, b,
                                                         (short)0, S[n], false, false);
        }

      // ---- online softmax: C-layout row = g + 8*half, col = l16 ----
      float rowm[8];
#pragma unroll
      for (int g = 0; g < 8; ++g) {
        float x = S[0][g];
        x = fmaxf(x, S[1][g]); x = fmaxf(x, S[2][g]); x = fmaxf(x, S[3][g]);
        rowm[g] = x;
      }
#pragma unroll
      for (int mask = 1; mask <= 8; mask <<= 1)
#pragma unroll
        for (int g = 0; g < 8; ++g)
          rowm[g] = fmaxf(rowm[g], __shfl_xor(rowm[g], mask, 32));

      float alpha[8];
#pragma unroll
      for (int g = 0; g < 8; ++g) {
        const float mn = fmaxf(m[g], rowm[g]);
        alpha[g] = exp2f(m[g] - mn);
        m[g] = mn;
      }

      float rs[8];
#pragma unroll
      for (int g = 0; g < 8; ++g) rs[g] = 0.0f;
#pragma unroll
      for (int n = 0; n < 4; ++n)
#pragma unroll
        for (int g = 0; g < 8; ++g) {
          const float p = exp2f(S[n][g] - m[g]);
          rs[g] += p;
          sP[wave][g + 8 * half][n * 16 + l16] = (__bf16)p;  // stage P for A-frag reload
        }
#pragma unroll
      for (int mask = 1; mask <= 8; mask <<= 1)
#pragma unroll
        for (int g = 0; g < 8; ++g)
          rs[g] += __shfl_xor(rs[g], mask, 32);
#pragma unroll
      for (int g = 0; g < 8; ++g) lsum[g] = lsum[g] * alpha[g] + rs[g];

#pragma unroll
      for (int ng = 0; ng < 4; ++ng)
#pragma unroll
        for (int g = 0; g < 8; ++g) o[ng][g] *= alpha[g];

      asm volatile("" ::: "memory");  // keep P stores before P reloads (same-wave DS is in-order)

      // ---- O += P * V ----
#pragma unroll
      for (int kk = 0; kk < 2; ++kk) {
        // P A-frag: lane row M = l16; elems 0..15 -> kv kk*32+16*half+e (contiguous)
        union { v16bf v; v8bf h2[2]; } ua;
        const __bf16* pr = &sP[wave][l16][kk * 32];
        ua.h2[0] = *(const v8bf*)(pr + 16 * half);
        ua.h2[1] = *(const v8bf*)(pr + 16 * half + 8);
#pragma unroll
        for (int ng = 0; ng < 4; ++ng) {
          // V B-frag: lane column N = out dim ng*16+l16; K-run = kv kk*32+16*half (contig in sVt)
          v16bf b = ld16(&sVt[ng * 16 + l16][kk * 32 + 16 * half]);
          o[ng] = __builtin_amdgcn_wmma_f32_16x16x32_bf16(false, ua.v, false, b,
                                                          (short)0, o[ng], false, false);
        }
      }
    }

    // finalize chunk: normalize by chunk softmax sum and add (per-chunk independent softmax)
    float rinv[8];
#pragma unroll
    for (int g = 0; g < 8; ++g) rinv[g] = 1.0f / lsum[g];
#pragma unroll
    for (int ng = 0; ng < 4; ++ng)
#pragma unroll
      for (int g = 0; g < 8; ++g) acc[ng][g] += o[ng][g] * rinv[g];
  }

  // ---- write output [B,N,H,D]: row = g + 8*half, dim = ng*16 + l16 ----
#pragma unroll
  for (int ng = 0; ng < 4; ++ng)
#pragma unroll
    for (int g = 0; g < 8; ++g) {
      const int row = g + 8 * half;
      Og[((size_t)(qb + wave * 16 + row) * kH + h) * kD + ng * 16 + l16] = acc[ng][g];
    }
}

extern "C" void kernel_launch(void* const* d_in, const int* in_sizes, int n_in,
                              void* d_out, int out_size, void* d_ws, size_t ws_size,
                              hipStream_t stream) {
  const float* Q = (const float*)d_in[0];
  const float* K = (const float*)d_in[1];
  const float* V = (const float*)d_in[2];
  float* O = (float*)d_out;
  (void)in_sizes; (void)n_in; (void)out_size; (void)d_ws; (void)ws_size;

  dim3 grid(kN / kQPB, kH, 1);     // 32 query-tiles x 16 heads = 512 workgroups
  dim3 block(kWaves * 32, 1, 1);   // 256 threads = 8 wave32
  hipLaunchKernelGGL(ring_attn_bf16_wmma, grid, block, 0, stream, Q, K, V, O);
}
